// GANN_86801289052223
// MI455X (gfx1250) — matrix operand order repaired and checked by reading, not compiled, not run on hardware
//
#include <hip/hip_runtime.h>
#include <hip/hip_bf16.h>

#define BB  4
#define NN  2048
#define KNN 20
#define PN  (BB*NN)        /* 8192   pixels for 1d convs  */
#define PEL (BB*NN*KNN)    /* 163840 pixels for edge convs */
#define NEGS 0.2f
#define EPSB 1e-5f

typedef _Float16 f16;
typedef __attribute__((ext_vector_type(16))) _Float16 v16h;
typedef __attribute__((ext_vector_type(8)))  _Float16 v8h;
typedef __attribute__((ext_vector_type(8)))  float    v8f;

// ---------------------------------------------------------------- utility
__global__ void col_mean_k(const float* __restrict__ x, float* __restrict__ mean3) {
  __shared__ float s[256];
  int bc = blockIdx.x;                       // b*3 + c, 12 blocks
  float acc = 0.f;
  for (int n = threadIdx.x; n < NN; n += 256) acc += x[(size_t)bc*NN + n];
  s[threadIdx.x] = acc; __syncthreads();
  for (int st = 128; st > 0; st >>= 1) { if (threadIdx.x < st) s[threadIdx.x] += s[threadIdx.x+st]; __syncthreads(); }
  if (threadIdx.x == 0) mean3[bc] = s[0] / (float)NN;
}

__global__ void repsurf_k(const float* __restrict__ x, const float* __restrict__ mean3,
                          float* __restrict__ f) {
  int t = blockIdx.x*blockDim.x + threadIdx.x;
  if (t >= BB*NN) return;
  int b = t / NN, n = t % NN;
  float p0 = x[((size_t)b*3+0)*NN+n], p1 = x[((size_t)b*3+1)*NN+n], p2 = x[((size_t)b*3+2)*NN+n];
  float r  = sqrtf(p0*p0 + p1*p1 + p2*p2);
  float* fb = f + (size_t)b*10*NN;
  fb[0*NN+n]=p0; fb[1*NN+n]=p1; fb[2*NN+n]=p2;
  fb[3*NN+n]=p0-mean3[b*3+0]; fb[4*NN+n]=p1-mean3[b*3+1]; fb[5*NN+n]=p2-mean3[b*3+2];
  fb[6*NN+n]=p0*p0; fb[7*NN+n]=p1*p1; fb[8*NN+n]=p2*p2;
  fb[9*NN+n]=r;
}

// ---------------------------------------------------------------- kNN (top-20, tie-stable)
__global__ void knn_k(const float* __restrict__ xin, int C, int* __restrict__ idx) {
  int t = blockIdx.x*blockDim.x + threadIdx.x;
  if (t >= BB*NN) return;
  int b = t / NN, n = t % NN;
  float q[128];
  for (int c = 0; c < C; ++c) q[c] = xin[((size_t)b*C + c)*NN + n];
  float bd[KNN]; int bi[KNN];
  for (int i = 0; i < KNN; ++i) { bd[i] = 1e30f; bi[i] = 0; }
  for (int m = 0; m < NN; ++m) {
    float d = 0.f;
    for (int c = 0; c < C; ++c) { float dx = q[c] - xin[((size_t)b*C + c)*NN + m]; d += dx*dx; }
    if (d < bd[KNN-1]) {
      int pos = KNN-1;
      while (pos > 0 && d < bd[pos-1]) { bd[pos]=bd[pos-1]; bi[pos]=bi[pos-1]; --pos; }
      bd[pos] = d; bi[pos] = m;
    }
  }
  for (int i = 0; i < KNN; ++i) idx[(size_t)t*KNN + i] = bi[i];
}

__global__ void nbmax_k(const float* __restrict__ xin, int C, const int* __restrict__ idx,
                        float* __restrict__ nbm) {
  size_t t = (size_t)blockIdx.x*blockDim.x + threadIdx.x;
  if (t >= (size_t)BB*C*NN) return;
  int n = t % NN; int c = (t / NN) % C; int b = t / ((size_t)C*NN);
  const int* ip = idx + ((size_t)b*NN + n)*KNN;
  float m = -1e30f;
  for (int k = 0; k < KNN; ++k) m = fmaxf(m, xin[((size_t)b*C + c)*NN + ip[k]]);
  nbm[((size_t)b*C + c)*NN + n] = m;
}

// rows [0,C)=x, [C,2C)=nbmax, rest zero-pad  (dst: Cfpad x PN, f16)
__global__ void fill_xf_k(const float* __restrict__ xin, const float* __restrict__ nbm,
                          int C, int Cfpad, f16* __restrict__ Xf) {
  size_t t = (size_t)blockIdx.x*blockDim.x + threadIdx.x;
  if (t >= (size_t)Cfpad*PN) return;
  int row = t / PN; int p = t % PN; int b = p / NN; int n = p % NN;
  float v = 0.f;
  if (row < C)        v = xin[((size_t)b*C + row)*NN + n];
  else if (row < 2*C) v = nbm[((size_t)b*C + (row-C))*NN + n];
  Xf[t] = (f16)v;
}

// ---------------------------------------------------------------- WMMA GEMM
// Y(Coutpad x P) = W(Coutpad x CINP) * X(CINP x P).
// 4 waves/block; each wave: one 16-row o-tile x 32 p-cols (2 accumulators, A reused).
// X tile (32K x 32P) staged transposed in LDS (bank-padded) -> ds_load_b128 fragments.
template<int CINP>
__global__ __launch_bounds__(128) void wmma_gemm_k(const f16* __restrict__ W,
                                                   const f16* __restrict__ X,
                                                   float* __restrict__ Y,
                                                   long long P) {
  constexpr int TSTR = 40;                         // padded stride (halves): 80B, 16B-aligned, bank-spread
  __shared__ __align__(16) f16 tileT[32][TSTR];    // [p-col][k-row]
  int tid  = threadIdx.x;
  int wave = tid >> 5, lane = tid & 31;
  int hi   = lane >> 4, M = lane & 15;
  int o0   = (blockIdx.y * 4 + wave) * 16;
  size_t pw = (size_t)blockIdx.x * 32;

  int lr = tid >> 2;                               // staged k-row (0..31)
  int lc = (tid & 3) * 8;                          // first staged p-col

  union { v16h v; v8h h[2]; f16 e[16]; } a, b0, b1;
  union { v8f  v; float e[8]; } acc0, acc1;
#pragma unroll
  for (int i = 0; i < 8; ++i) { acc0.e[i] = 0.f; acc1.e[i] = 0.f; }

  // A 16x32 f16 k order used here (same as B): lanes 0-15 hold K {0..15}, lanes 16-31 hold {16..31}
  const f16* wrow = W + (size_t)(o0 + M) * CINP + (hi ? 16 : 0);

  for (int kk = 0; kk < CINP; kk += 32) {
    v8h xv = *(const v8h*)(X + (size_t)(kk + lr) * P + pw + lc);   // coalesced b128
    if (kk + 32 < CINP)
      __builtin_prefetch(X + (size_t)(kk + 32 + lr) * P + pw + lc, 0, 1);
#pragma unroll
    for (int i = 0; i < 8; ++i) tileT[lc + i][lr] = xv[i];         // transpose scatter
    __syncthreads();

    a.h[0]  = *(const v8h*)(wrow + kk);
    a.h[1]  = *(const v8h*)(wrow + kk + 8);
    b0.h[0] = *(const v8h*)&tileT[M     ][hi * 16];                // 16 consecutive K rows
    b0.h[1] = *(const v8h*)&tileT[M     ][hi * 16 + 8];
    b1.h[0] = *(const v8h*)&tileT[M + 16][hi * 16];
    b1.h[1] = *(const v8h*)&tileT[M + 16][hi * 16 + 8];

    acc0.v = __builtin_amdgcn_wmma_f32_16x16x32_f16(false, a.v, false, b0.v,
                                                    (short)0, acc0.v, false, false);
    acc1.v = __builtin_amdgcn_wmma_f32_16x16x32_f16(false, a.v, false, b1.v,
                                                    (short)0, acc1.v, false, false);
    __syncthreads();
  }
#pragma unroll
  for (int r = 0; r < 8; ++r) {                    // D: VGPR r -> M = r (+8 for hi lanes)
    size_t row = (size_t)(o0 + (hi ? 8 : 0) + r) * P;
    Y[row + pw + M]      = acc0.e[r];
    Y[row + pw + 16 + M] = acc1.e[r];
  }
}

// ---------------------------------------------------------------- stage glue
__global__ void fill_gf_k(const float* __restrict__ Yfn, f16* __restrict__ Xe, int rowOff) {
  size_t t = (size_t)blockIdx.x*blockDim.x + threadIdx.x;
  if (t >= (size_t)512*PN) return;
  int o = t / PN; int p = t % PN;
  f16 v = (f16)fmaxf(Yfn[(size_t)o*PN + p], 0.f);
  f16* dst = Xe + (size_t)(rowOff + o)*PEL + (size_t)p*KNN;
  for (int k = 0; k < KNN; ++k) dst[k] = v;
}

__global__ void zero_f16_k(f16* __restrict__ X, size_t count) {
  size_t t = (size_t)blockIdx.x*blockDim.x + threadIdx.x;
  if (t < count) X[t] = (f16)0.f;
}

// wave-per-pixel attention: writes Xe rows [0,C)=alpha*diff, [C,2C)=x
__global__ __launch_bounds__(128) void attn_edge_k(const float* __restrict__ xin,
                                                   const int* __restrict__ idx,
                                                   const float* __restrict__ Wa1,
                                                   const float* __restrict__ Wa2,
                                                   f16* __restrict__ Xe, int C) {
  int wave = threadIdx.x >> 5, lane = threadIdx.x & 31;
  size_t pe = (size_t)blockIdx.x*4 + wave;            // grid exact: PEL/4 blocks
  int k = pe % KNN; int n = (pe / KNN) % NN; int b = pe / ((size_t)KNN*NN);
  __shared__ float sdiff[4][128];
  __shared__ float sxv[4][128];
  int j = idx[((size_t)b*NN + n)*KNN + k];
  for (int c = lane; c < C; c += 32) {
    float xv = xin[((size_t)b*C + c)*NN + n];
    float nb = xin[((size_t)b*C + c)*NN + j];
    sxv[wave][c]  = xv;
    sdiff[wave][c] = nb - xv;
  }
  __syncthreads();
  float aval[8]; int cnt = 0;
  float mx = -1e30f;
  for (int c = lane; c < C; c += 32) {
    float t = 0.f;
    for (int c2 = 0; c2 < C; ++c2)
      t += Wa1[c*C + c2]*sdiff[wave][c2] + Wa2[c*C + c2]*sxv[wave][c2];
    t = (t > 0.f) ? t : NEGS*t;                        // leaky_relu
    aval[cnt++] = t;
    mx = fmaxf(mx, t);
  }
  for (int off = 16; off > 0; off >>= 1) mx = fmaxf(mx, __shfl_xor(mx, off, 32));
  float s = 0.f;
  for (int i = 0; i < cnt; ++i) { aval[i] = expf(aval[i] - mx); s += aval[i]; }
  for (int off = 16; off > 0; off >>= 1) s += __shfl_xor(s, off, 32);
  float inv = 1.f / s;
  cnt = 0;
  for (int c = lane; c < C; c += 32) {
    float alpha = aval[cnt++] * inv;
    Xe[(size_t)c*PEL + pe]       = (f16)(alpha * sdiff[wave][c]);
    Xe[(size_t)(C + c)*PEL + pe] = (f16)sxv[wave][c];
  }
}

// ---------------------------------------------------------------- BN / reductions
__global__ void bn_stats_k(const float* __restrict__ Y, long long P,
                           float* __restrict__ m, float* __restrict__ v) {
  __shared__ float ss[256], sq[256];
  size_t base = (size_t)blockIdx.x * P;
  float s = 0.f, q = 0.f;
  for (long long p = threadIdx.x; p < P; p += 256) { float t = Y[base + p]; s += t; q += t*t; }
  ss[threadIdx.x] = s; sq[threadIdx.x] = q; __syncthreads();
  for (int st = 128; st > 0; st >>= 1) {
    if (threadIdx.x < st) { ss[threadIdx.x] += ss[threadIdx.x+st]; sq[threadIdx.x] += sq[threadIdx.x+st]; }
    __syncthreads();
  }
  if (threadIdx.x == 0) { float mm = ss[0]/(float)P; m[blockIdx.x] = mm; v[blockIdx.x] = sq[0]/(float)P - mm*mm; }
}

__global__ void bn_relu_maxk_k(const float* __restrict__ Ye, const float* __restrict__ m,
                               const float* __restrict__ v, const float* __restrict__ g,
                               const float* __restrict__ bb, float* __restrict__ xout, int Cout) {
  size_t t = (size_t)blockIdx.x*blockDim.x + threadIdx.x;
  if (t >= (size_t)BB*Cout*NN) return;
  int n = t % NN; int o = (t / NN) % Cout; int b = t / ((size_t)Cout*NN);
  float sc = g[o] * rsqrtf(v[o] + EPSB);
  float sh = bb[o] - m[o]*sc;
  size_t base = (size_t)o*PEL + ((size_t)b*NN + n)*KNN;
  float mx = -1e30f;
  for (int k = 0; k < KNN; ++k) mx = fmaxf(mx, Ye[base + k]*sc + sh);
  xout[((size_t)b*Cout + o)*NN + n] = fmaxf(mx, 0.f);   // relu commutes with max
}

__global__ void copy_rows_k(const float* __restrict__ src, int C, int rowOff, f16* __restrict__ dst) {
  size_t t = (size_t)blockIdx.x*blockDim.x + threadIdx.x;
  if (t >= (size_t)BB*C*NN) return;
  int n = t % NN; int c = (t / NN) % C; int b = t / ((size_t)C*NN);
  dst[(size_t)(rowOff + c)*PN + (size_t)b*NN + n] = (f16)src[t];
}

__global__ void bn_relu_f32_k(const float* __restrict__ Y, const float* __restrict__ m,
                              const float* __restrict__ v, const float* __restrict__ g,
                              const float* __restrict__ bb, float* __restrict__ H, int Cout) {
  size_t t = (size_t)blockIdx.x*blockDim.x + threadIdx.x;
  if (t >= (size_t)Cout*PN) return;
  int o = t / PN;
  float sc = g[o] * rsqrtf(v[o] + EPSB);
  H[t] = fmaxf(Y[t]*sc + (bb[o] - m[o]*sc), 0.f);
}

__global__ void gmax_k(const float* __restrict__ H, float* __restrict__ gvec) {
  int t = blockIdx.x*blockDim.x + threadIdx.x;
  if (t >= BB*1024) return;
  int b = t / 1024, o = t % 1024;
  float mx = -1e30f;
  for (int n = 0; n < NN; ++n) mx = fmaxf(mx, H[(size_t)o*PN + (size_t)b*NN + n]);
  gvec[t] = mx;
}

__global__ void fill_gvec_k(const float* __restrict__ gvec, f16* __restrict__ X6) {
  size_t t = (size_t)blockIdx.x*blockDim.x + threadIdx.x;
  if (t >= (size_t)1024*PN) return;
  int o = t / PN; int p = t % PN; int b = p / NN;
  X6[(size_t)(128 + o)*PN + p] = (f16)gvec[b*1024 + o];
}

__global__ void bn_relu_f16_k(const float* __restrict__ Y, const float* __restrict__ m,
                              const float* __restrict__ v, const float* __restrict__ g,
                              const float* __restrict__ bb, f16* __restrict__ Xd, int Cout) {
  size_t t = (size_t)blockIdx.x*blockDim.x + threadIdx.x;
  if (t >= (size_t)Cout*PN) return;
  int o = t / PN;
  float sc = g[o] * rsqrtf(v[o] + EPSB);
  Xd[t] = (f16)fmaxf(Y[t]*sc + (bb[o] - m[o]*sc), 0.f);
}

__global__ void relu_out_k(const float* __restrict__ Y8, float* __restrict__ out) {
  size_t t = (size_t)blockIdx.x*blockDim.x + threadIdx.x;
  if (t >= (size_t)BB*50*NN) return;
  int n = t % NN; int o = (t / NN) % 50; int b = t / (50*NN);
  out[t] = fmaxf(Y8[(size_t)o*PN + (size_t)b*NN + n], 0.f);
}

__global__ void pack_w_k(const float* __restrict__ W, f16* __restrict__ Wp,
                         int Cout, int Cin, int Cinp, size_t total) {
  size_t t = (size_t)blockIdx.x*blockDim.x + threadIdx.x;
  if (t >= total) return;
  int o = t / Cinp, c = t % Cinp;
  Wp[t] = (o < Cout && c < Cin) ? (f16)W[(size_t)o*Cin + c] : (f16)0.f;
}

// ================================================================ host
static inline unsigned cdiv(size_t a, unsigned b) { return (unsigned)((a + b - 1) / b); }

static void gemm(const f16* W, const f16* X, float* Y, int Coutpad, int Cinpad,
                 long long P, hipStream_t s) {
  dim3 grid((unsigned)(P/32), (unsigned)(Coutpad/64));
  switch (Cinpad) {
    case 32:   wmma_gemm_k<32>  <<<grid,128,0,s>>>(W,X,Y,P); break;
    case 128:  wmma_gemm_k<128> <<<grid,128,0,s>>>(W,X,Y,P); break;
    case 256:  wmma_gemm_k<256> <<<grid,128,0,s>>>(W,X,Y,P); break;
    case 512:  wmma_gemm_k<512> <<<grid,128,0,s>>>(W,X,Y,P); break;
    case 544:  wmma_gemm_k<544> <<<grid,128,0,s>>>(W,X,Y,P); break;
    case 640:  wmma_gemm_k<640> <<<grid,128,0,s>>>(W,X,Y,P); break;
    case 768:  wmma_gemm_k<768> <<<grid,128,0,s>>>(W,X,Y,P); break;
    case 1152: wmma_gemm_k<1152><<<grid,128,0,s>>>(W,X,Y,P); break;
    default: break;
  }
}

extern "C" void kernel_launch(void* const* d_in, const int* in_sizes, int n_in,
                              void* d_out, int out_size, void* d_ws, size_t ws_size,
                              hipStream_t stream) {
  const float* x = (const float*)d_in[0];
  const float* wa[8]; for (int i = 0; i < 8; ++i) wa[i] = (const float*)d_in[1+i];
  const float* g[7];  const float* bt[7];
  for (int i = 0; i < 7; ++i) { g[i] = (const float*)d_in[21 + 2*i]; bt[i] = (const float*)d_in[22 + 2*i]; }

  char* ws = (char*)d_ws;
  size_t off = 0;
  auto carve = [&](size_t bytes) -> void* {
    void* r = ws + off;
    off += (bytes + 255) & ~(size_t)255;
    return r;
  };
  float* f     = (float*)carve((size_t)BB*10*NN*4);
  float* x1    = (float*)carve((size_t)BB*64*NN*4);
  float* x2    = (float*)carve((size_t)BB*64*NN*4);
  float* x3    = (float*)carve((size_t)BB*128*NN*4);
  float* x4    = (float*)carve((size_t)BB*256*NN*4);
  float* mean3 = (float*)carve(12*4);
  int*   idxb  = (int*)  carve((size_t)BB*NN*KNN*4);
  float* nbm   = (float*)carve((size_t)BB*128*NN*4);
  f16*   Xf    = (f16*)  carve((size_t)256*PN*2);
  float* Yfn   = (float*)carve((size_t)512*PN*4);
  float* mbuf  = (float*)carve(1024*4);
  float* vbuf  = (float*)carve(1024*4);
  float* gvec  = (float*)carve((size_t)BB*1024*4);

  // packed f16 weights: {src, Cout, Cin, Coutpad, Cinpad}
  const int WT[12][5] = {
    { 9, 512,  20, 512,  32}, {10, 512, 128, 512, 128}, {11, 512, 128, 512, 128}, {12, 512, 256, 512, 256},
    {13,  64, 532,  64, 544}, {14,  64, 640,  64, 640}, {15, 128, 640, 128, 640}, {16, 256, 768, 256, 768},
    {17,1024, 512,1024, 512}, {18, 512,1152, 512,1152}, {19, 256, 512, 256, 512}, {20,  50, 256,  64, 256}};
  f16* Wp[12];
  for (int i = 0; i < 12; ++i) Wp[i] = (f16*)carve((size_t)WT[i][3]*WT[i][4]*2);

  char* XeB = (char*)carve((size_t)768*PEL*2);   // edge-conv activations (also hosts head X buffers)
  char* YeB = (char*)carve((size_t)256*PEL*4);   // edge-conv outputs   (also hosts head Y buffers)
  f16*   Xe = (f16*)XeB;
  float* Ye = (float*)YeB;
  f16*   Xc = (f16*)(XeB);                          // 512 x 8192
  f16*   X6 = (f16*)(XeB + (size_t)16*1024*1024);   // 1152 x 8192
  f16*   X7 = (f16*)(XeB + (size_t)48*1024*1024);   // 512 x 8192
  f16*   X8 = (f16*)(XeB + (size_t)64*1024*1024);   // 256 x 8192
  float* Y5 = (float*)(YeB);                        // 1024 x 8192
  float* H5 = (float*)(YeB + (size_t)34*1024*1024); // 1024 x 8192
  float* Y6 = (float*)(YeB + (size_t)72*1024*1024); // 512 x 8192
  float* Y7 = (float*)(YeB + (size_t)96*1024*1024); // 256 x 8192
  float* Y8 = (float*)(YeB + (size_t)112*1024*1024);// 64 x 8192
  (void)ws_size; (void)n_in; (void)in_sizes; (void)out_size;

  // pack weights
  for (int i = 0; i < 12; ++i) {
    size_t tot = (size_t)WT[i][3]*WT[i][4];
    pack_w_k<<<cdiv(tot,256),256,0,stream>>>((const float*)d_in[WT[i][0]], Wp[i],
                                             WT[i][1], WT[i][2], WT[i][4], tot);
  }

  // repsurf
  col_mean_k<<<12,256,0,stream>>>(x, mean3);
  repsurf_k<<<cdiv((size_t)BB*NN,256),256,0,stream>>>(x, mean3, f);

  auto stage = [&](const float* xin, int C, int Cout, const float* Wa1, const float* Wa2,
                   const f16* Wfn, int CinFp, const f16* Wec, int CinEp,
                   const float* gg, const float* bb, float* xout) {
    knn_k  <<<cdiv((size_t)BB*NN,128),128,0,stream>>>(xin, C, idxb);
    nbmax_k<<<cdiv((size_t)BB*C*NN,256),256,0,stream>>>(xin, C, idxb, nbm);
    fill_xf_k<<<cdiv((size_t)CinFp*PN,256),256,0,stream>>>(xin, nbm, C, CinFp, Xf);
    gemm(Wfn, Xf, Yfn, 512, CinFp, PN, stream);
    fill_gf_k<<<cdiv((size_t)512*PN,256),256,0,stream>>>(Yfn, Xe, 2*C);
    int gfEnd = 2*C + 512;
    if (CinEp > gfEnd) {
      size_t cnt = (size_t)(CinEp - gfEnd)*PEL;
      zero_f16_k<<<cdiv(cnt,256),256,0,stream>>>(Xe + (size_t)gfEnd*PEL, cnt);
    }
    attn_edge_k<<<PEL/4,128,0,stream>>>(xin, idxb, Wa1, Wa2, Xe, C);
    gemm(Wec, Xe, Ye, Cout, CinEp, PEL, stream);
    bn_stats_k<<<Cout,256,0,stream>>>(Ye, (long long)PEL, mbuf, vbuf);
    bn_relu_maxk_k<<<cdiv((size_t)BB*Cout*NN,256),256,0,stream>>>(Ye, mbuf, vbuf, gg, bb, xout, Cout);
  };

  stage(f , 10 ,  64, wa[0], wa[1], Wp[0],  32, Wp[4], 544, g[0], bt[0], x1);
  stage(x1, 64 ,  64, wa[2], wa[3], Wp[1], 128, Wp[5], 640, g[1], bt[1], x2);
  stage(x2, 64 , 128, wa[4], wa[5], Wp[2], 128, Wp[6], 640, g[2], bt[2], x3);
  stage(x3, 128, 256, wa[6], wa[7], Wp[3], 256, Wp[7], 768, g[3], bt[3], x4);

  // head: wec5 on cat(x1,x2,x3,x4)
  copy_rows_k<<<cdiv((size_t)BB*64*NN,256),256,0,stream>>>(x1,  64,   0, Xc);
  copy_rows_k<<<cdiv((size_t)BB*64*NN,256),256,0,stream>>>(x2,  64,  64, Xc);
  copy_rows_k<<<cdiv((size_t)BB*128*NN,256),256,0,stream>>>(x3, 128, 128, Xc);
  copy_rows_k<<<cdiv((size_t)BB*256*NN,256),256,0,stream>>>(x4, 256, 256, Xc);
  gemm(Wp[8], Xc, Y5, 1024, 512, PN, stream);
  bn_stats_k<<<1024,256,0,stream>>>(Y5, (long long)PN, mbuf, vbuf);
  bn_relu_f32_k<<<cdiv((size_t)1024*PN,256),256,0,stream>>>(Y5, mbuf, vbuf, g[4], bt[4], H5, 1024);
  gmax_k<<<cdiv((size_t)BB*1024,256),256,0,stream>>>(H5, gvec);

  // wec6 on cat(x3, gvec)
  copy_rows_k<<<cdiv((size_t)BB*128*NN,256),256,0,stream>>>(x3, 128, 0, X6);
  fill_gvec_k<<<cdiv((size_t)1024*PN,256),256,0,stream>>>(gvec, X6);
  gemm(Wp[9], X6, Y6, 512, 1152, PN, stream);
  bn_stats_k<<<512,256,0,stream>>>(Y6, (long long)PN, mbuf, vbuf);
  bn_relu_f16_k<<<cdiv((size_t)512*PN,256),256,0,stream>>>(Y6, mbuf, vbuf, g[5], bt[5], X7, 512);

  gemm(Wp[10], X7, Y7, 256, 512, PN, stream);
  bn_stats_k<<<256,256,0,stream>>>(Y7, (long long)PN, mbuf, vbuf);
  bn_relu_f16_k<<<cdiv((size_t)256*PN,256),256,0,stream>>>(Y7, mbuf, vbuf, g[6], bt[6], X8, 256);

  gemm(Wp[11], X8, Y8, 64, 256, PN, stream);
  relu_out_k<<<cdiv((size_t)BB*50*NN,256),256,0,stream>>>(Y8, (float*)d_out);
}
